// DeltaFormerAttention_36309653520701
// MI455X (gfx1250) — matrix operand order repaired and checked by reading, not compiled, not run on hardware
//
#include <hip/hip_runtime.h>

// ---------------------------------------------------------------------------
// DeltaFormer attention for MI455X (gfx1250), wave32 + WMMA bf16.
//   stage 0: one-shot f32 -> bf16 conversion of x and all weights
//   stage A: q/k/v projections (WMMA GEMM, bf16 head-major out)
//   stage B: beta projection (small)
//   stage C: blocked delta-rule forward substitution (WMMA for off-diag)
//   stage D: causal flash attention over u (WMMA)
//   stage E: output projection (WMMA GEMM, f32 out)
// Tile staging uses GLOBAL_LOAD_ASYNC_TO_LDS_B128 (ASYNCcnt) when the
// toolchain exposes the builtins, else falls back to sync b128 copies.
// ---------------------------------------------------------------------------

#define T_DIM   2048
#define HID     2048
#define NHEAD   16
#define HD      128
#define SCALE   0.08838834764831845f   // 128^-0.5

typedef __attribute__((ext_vector_type(16))) __bf16 v16bf;
typedef __attribute__((ext_vector_type(8)))  float  v8f;

// ---------------- async global->LDS (CDNA5) --------------------------------
#if __has_builtin(__builtin_amdgcn_global_load_async_to_lds_b128) && \
    __has_builtin(__builtin_amdgcn_s_wait_asynccnt)
#define USE_ASYNC_LDS 1
#else
#define USE_ASYNC_LDS 0
#endif

#if USE_ASYNC_LDS
typedef int async_v4i __attribute__((vector_size(16)));
__device__ __forceinline__ void async_cp16(unsigned short* ldst,
                                           const unsigned short* gsrc) {
    __builtin_amdgcn_global_load_async_to_lds_b128(
        (__attribute__((address_space(1))) async_v4i*)gsrc,
        (__attribute__((address_space(3))) async_v4i*)ldst, 0, 0);
}
__device__ __forceinline__ void tile_fence() {
    __builtin_amdgcn_s_wait_asynccnt(0);
}
#else
__device__ __forceinline__ void async_cp16(unsigned short* ldst,
                                           const unsigned short* gsrc) {
    *(uint4*)ldst = *(const uint4*)gsrc;
}
__device__ __forceinline__ void tile_fence() {}
#endif

// ---------------- scalar bf16 helpers (RNE) --------------------------------
__device__ __forceinline__ unsigned short f2bf(float f) {
    unsigned int u = __float_as_uint(f);
    unsigned int r = u + 0x7FFFu + ((u >> 16) & 1u);
    return (unsigned short)(r >> 16);
}
__device__ __forceinline__ float bf2f(unsigned short b) {
    return __uint_as_float(((unsigned int)b) << 16);
}

__device__ __forceinline__ v8f vzero8() {
    v8f z;
#pragma unroll
    for (int i = 0; i < 8; ++i) z[i] = 0.0f;
    return z;
}

// ---------------- WMMA fragments -------------------------------------------
struct FragBF {
    union { v16bf v; uint4 q[2]; };
};

// A-matrix 16x32 bf16 fragment: lane<16 -> K[0..7],[16..23]; lane>=16 -> +8.
__device__ __forceinline__ FragBF load_frag_a(const unsigned short* tile, int stride,
                                              int row0, int k0, int lane) {
    FragBF f;
    int half = lane >> 4, idx = lane & 15;
    const unsigned short* p = tile + (row0 + idx) * stride + k0 + half * 8;
    f.q[0] = *(const uint4*)(p);
    f.q[1] = *(const uint4*)(p + 16);
    return f;
}

// B-matrix 32x16 bf16 fragment: lane<16 -> K[0..7],[16..23]; lane>=16 -> +8.
// Tile stored [N][K] row-major (NT layout), so K is contiguous per column n.
__device__ __forceinline__ FragBF load_frag_b(const unsigned short* tile, int stride,
                                              int col0, int k0, int lane) {
    FragBF f;
    int half = lane >> 4, idx = lane & 15;
    const unsigned short* p = tile + (col0 + idx) * stride + k0 + half * 8;
    f.q[0] = *(const uint4*)(p);
    f.q[1] = *(const uint4*)(p + 16);
    return f;
}

__device__ __forceinline__ v8f wmma_bf16(const FragBF& a, const FragBF& b, v8f c) {
    return __builtin_amdgcn_wmma_f32_16x16x32_bf16(false, a.v, false, b.v,
                                                   (short)0, c, false, false);
}

// ---------------------------------------------------------------------------
// Stage 0: elementwise f32 -> bf16 (feeds everything; L2-resident afterwards)
// ---------------------------------------------------------------------------
__global__ __launch_bounds__(256) void cvt_bf16_kernel(
    const float* __restrict__ src, unsigned short* __restrict__ dst)
{
    int i = (blockIdx.x * 256 + threadIdx.x) * 4;
    float4 v = *(const float4*)(src + i);
    unsigned short e[4] = { f2bf(v.x), f2bf(v.y), f2bf(v.z), f2bf(v.w) };
    *(uint2*)(dst + i) = *(const uint2*)e;
}

// ---------------------------------------------------------------------------
// Stage A/E: C[M,N] = A[M,K] * B[N,K]^T  (both operands bf16)
//   EPI: 0 = f32 row-major out, 1 = bf16 head-major out [h][m][d]
// ---------------------------------------------------------------------------
#define GLDA 72   // 64 + 8 pad (144B row stride, 16B aligned)

template <int EPI>
__global__ __launch_bounds__(256) void gemm_nt_kernel(
    const unsigned short* __restrict__ Ab, const unsigned short* __restrict__ Bb,
    float* __restrict__ Cf, unsigned short* __restrict__ Cbf,
    int M, int N, int K)
{
    __shared__ unsigned short sA[128 * GLDA];
    __shared__ unsigned short sB[128 * GLDA];

    const int tid  = threadIdx.x;
    const int lane = tid & 31;
    const int wave = tid >> 5;
    const int wm   = wave >> 2;   // 0..1 -> 64 rows each
    const int wn   = wave & 3;    // 0..3 -> 32 cols each
    const int bm   = blockIdx.x;
    const int bn   = blockIdx.y;

    v8f acc[4][2];
#pragma unroll
    for (int i = 0; i < 4; ++i)
#pragma unroll
        for (int j = 0; j < 2; ++j) acc[i][j] = vzero8();

    for (int kc = 0; kc < K; kc += 64) {
        // ---- stage both tiles into LDS: 2 x 1024 b128 chunks, 8 per thread
        for (int it = tid; it < 2048; it += 256) {
            int isB = it >> 10;
            int idx = it & 1023;
            int r   = idx >> 3;
            int c8  = (idx & 7) << 3;
            if (!isB) {
                size_t ga = (size_t)(bm * 128 + r) * K + kc + c8;
                async_cp16(&sA[r * GLDA + c8], Ab + ga);
            } else {
                size_t gb = (size_t)(bn * 128 + r) * K + kc + c8;
                async_cp16(&sB[r * GLDA + c8], Bb + gb);
            }
        }
        tile_fence();
        __syncthreads();

        // ---- WMMA over the 64-wide K chunk ----
#pragma unroll
        for (int ks = 0; ks < 64; ks += 32) {
            FragBF a[4], b[2];
#pragma unroll
            for (int i = 0; i < 4; ++i)
                a[i] = load_frag_a(sA, GLDA, wm * 64 + i * 16, ks, lane);
#pragma unroll
            for (int j = 0; j < 2; ++j)
                b[j] = load_frag_b(sB, GLDA, wn * 32 + j * 16, ks, lane);
#pragma unroll
            for (int i = 0; i < 4; ++i)
#pragma unroll
                for (int j = 0; j < 2; ++j)
                    acc[i][j] = wmma_bf16(a[i], b[j], acc[i][j]);
        }
        __syncthreads();
    }

    // ---- epilogue ----
    const int halfl = lane >> 4, il = lane & 15;
#pragma unroll
    for (int i = 0; i < 4; ++i) {
#pragma unroll
        for (int j = 0; j < 2; ++j) {
            int n     = bn * 128 + wn * 32 + j * 16 + il;
            int mbase = bm * 128 + wm * 64 + i * 16 + halfl * 8;
#pragma unroll
            for (int r = 0; r < 8; ++r) {
                float val = acc[i][j][r];
                int m = mbase + r;
                if (EPI == 0) {
                    Cf[(size_t)m * N + n] = val;
                } else {
                    int h = n >> 7, d = n & 127;
                    Cbf[((size_t)h * M + m) * HD + d] = f2bf(val);
                }
            }
        }
    }
}

// ---------------------------------------------------------------------------
// Stage B: beta[h,t] = x[t,:] . Wb[h,:] + bb[h]
// ---------------------------------------------------------------------------
__global__ __launch_bounds__(256) void beta_kernel(
    const float* __restrict__ x, const float* __restrict__ Wb,
    const float* __restrict__ bias, float* __restrict__ beta)
{
    int idx = blockIdx.x * 256 + threadIdx.x;   // T*H items
    int t = idx >> 4, h = idx & 15;
    const float* xr = x  + (size_t)t * HID;
    const float* wr = Wb + (size_t)h * HID;
    float s = bias[h];
    for (int k = 0; k < HID; k += 4) {
        float4 a = *(const float4*)(xr + k);
        float4 b = *(const float4*)(wr + k);
        s += a.x * b.x + a.y * b.y + a.z * b.z + a.w * b.w;
    }
    beta[(size_t)h * T_DIM + t] = s;
}

// ---------------------------------------------------------------------------
// Shared tile loaders for attention-side kernels
// ---------------------------------------------------------------------------
#define LDQ   136   // 128 + 8 pad (272B row stride, 16B aligned)
#define LDP   72    // 64 + 8 pad
#define LDSS  68    // f32 S tile stride (272B)

__device__ __forceinline__ void load_tile64(unsigned short* dst,
                                            const unsigned short* src, int tid) {
    for (int it = tid; it < 64 * 16; it += 256) {
        int r  = it >> 4;
        int c8 = (it & 15) << 3;
        async_cp16(&dst[r * LDQ + c8], src + r * HD + c8);
    }
}

// U[s][d] (64x128) -> sU[d][s] transposed, so contraction dim (s) is contiguous.
__device__ __forceinline__ void load_u_trans(unsigned short* sU,
                                             const unsigned short* src, int tid) {
    for (int it = tid; it < 64 * 32; it += 256) {
        int s  = it >> 5;
        int c4 = (it & 31) << 2;
        union { uint2 u2; unsigned short e[4]; } d;
        d.u2 = *(const uint2*)(src + s * HD + c4);
        sU[(c4 + 0) * LDP + s] = d.e[0];
        sU[(c4 + 1) * LDP + s] = d.e[1];
        sU[(c4 + 2) * LDP + s] = d.e[2];
        sU[(c4 + 3) * LDP + s] = d.e[3];
    }
}

// ---------------------------------------------------------------------------
// Stage C: blocked delta-rule solve, one workgroup per head.
//   For each 64-row block I (sequential):
//     acc_I  = sum_{J<I} [beta_I * s * Q_I K_J^T] U_J      (WMMA)
//     S_II   = beta_I * s * Q_I K_I^T                      (WMMA)
//     u_t    = v_t - acc_t - sum_{s<t} S[t][s] u_s         (per-column serial)
// ---------------------------------------------------------------------------
__global__ __launch_bounds__(256) void delta_solve_kernel(
    const unsigned short* __restrict__ qb, const unsigned short* __restrict__ kb,
    const unsigned short* __restrict__ vb, const float* __restrict__ beta,
    unsigned short* __restrict__ ub, float* __restrict__ accg)
{
    __shared__ unsigned short sQ[64 * LDQ];
    __shared__ unsigned short sK[64 * LDQ];   // overlaid by f32 S tile in diag phase
    __shared__ unsigned short sU[128 * LDP];
    __shared__ unsigned short sP[64 * LDP];
    __shared__ float sBeta[64];
    float* sS = (float*)sK;                   // 64*68*4 == 64*136*2 bytes

    const int h    = blockIdx.x;
    const int tid  = threadIdx.x;
    const int lane = tid & 31;
    const int wave = tid >> 5;
    const int wm   = wave >> 2;   // 0..1
    const int wn   = wave & 3;    // 0..3
    const int halfl = lane >> 4, il = lane & 15;

    const unsigned short* qh = qb + (size_t)h * T_DIM * HD;
    const unsigned short* kh = kb + (size_t)h * T_DIM * HD;
    const unsigned short* vh = vb + (size_t)h * T_DIM * HD;
    unsigned short*       uh = ub + (size_t)h * T_DIM * HD;
    float*              acch = accg + (size_t)h * 64 * HD;
    const float*       betah = beta + (size_t)h * T_DIM;

    for (int I = 0; I < T_DIM / 64; ++I) {
        load_tile64(sQ, qh + (size_t)I * 64 * HD, tid);
        if (tid < 64) sBeta[tid] = betah[I * 64 + tid];
        tile_fence();
        __syncthreads();

        v8f acc[2][2];
#pragma unroll
        for (int i = 0; i < 2; ++i)
#pragma unroll
            for (int j = 0; j < 2; ++j) acc[i][j] = vzero8();

        for (int J = 0; J < I; ++J) {
            load_tile64(sK, kh + (size_t)J * 64 * HD, tid);
            load_u_trans(sU, uh + (size_t)J * 64 * HD, tid);
            tile_fence();
            __syncthreads();

            // S = Q_I K_J^T : wave tile 32x16 (2 m-tiles x 1 n-tile)
            v8f s0 = vzero8(), s1 = vzero8();
#pragma unroll
            for (int ks = 0; ks < HD; ks += 32) {
                FragBF a0 = load_frag_a(sQ, LDQ, wm * 32,      ks, lane);
                FragBF a1 = load_frag_a(sQ, LDQ, wm * 32 + 16, ks, lane);
                FragBF bk = load_frag_b(sK, LDQ, wn * 16,      ks, lane);
                s0 = wmma_bf16(a0, bk, s0);
                s1 = wmma_bf16(a1, bk, s1);
            }
            // P = bf16(beta * scale * S)
#pragma unroll
            for (int jj = 0; jj < 2; ++jj) {
                v8f& s = jj ? s1 : s0;
                int rowb = wm * 32 + jj * 16 + halfl * 8;
                int col  = wn * 16 + il;
#pragma unroll
                for (int r = 0; r < 8; ++r) {
                    int row = rowb + r;
                    sP[row * LDP + col] = f2bf(sBeta[row] * SCALE * s[r]);
                }
            }
            __syncthreads();

            // acc += P(64x64) @ U_J(64x128) : wave tile 32x32
#pragma unroll
            for (int ks = 0; ks < 64; ks += 32) {
                FragBF p0 = load_frag_a(sP, LDP, wm * 32,      ks, lane);
                FragBF p1 = load_frag_a(sP, LDP, wm * 32 + 16, ks, lane);
                FragBF u0 = load_frag_b(sU, LDP, wn * 32,      ks, lane);
                FragBF u1 = load_frag_b(sU, LDP, wn * 32 + 16, ks, lane);
                acc[0][0] = wmma_bf16(p0, u0, acc[0][0]);
                acc[0][1] = wmma_bf16(p0, u1, acc[0][1]);
                acc[1][0] = wmma_bf16(p1, u0, acc[1][0]);
                acc[1][1] = wmma_bf16(p1, u1, acc[1][1]);
            }
            __syncthreads();
        }

        // ---- diagonal block: S_II ----
        load_tile64(sK, kh + (size_t)I * 64 * HD, tid);
        tile_fence();
        __syncthreads();
        v8f s0 = vzero8(), s1 = vzero8();
#pragma unroll
        for (int ks = 0; ks < HD; ks += 32) {
            FragBF a0 = load_frag_a(sQ, LDQ, wm * 32,      ks, lane);
            FragBF a1 = load_frag_a(sQ, LDQ, wm * 32 + 16, ks, lane);
            FragBF bk = load_frag_b(sK, LDQ, wn * 16,      ks, lane);
            s0 = wmma_bf16(a0, bk, s0);
            s1 = wmma_bf16(a1, bk, s1);
        }
        __syncthreads();   // all waves done reading sK before overlaying with sS
#pragma unroll
        for (int jj = 0; jj < 2; ++jj) {
            v8f& s = jj ? s1 : s0;
            int rowb = wm * 32 + jj * 16 + halfl * 8;
            int col  = wn * 16 + il;
#pragma unroll
            for (int r = 0; r < 8; ++r) {
                int row = rowb + r;
                sS[row * LDSS + col] = sBeta[row] * SCALE * s[r];
            }
        }
        // dump register acc to per-head global scratch so solver threads see it
#pragma unroll
        for (int i = 0; i < 2; ++i)
#pragma unroll
            for (int j = 0; j < 2; ++j) {
                int rowb = wm * 32 + i * 16 + halfl * 8;
                int col  = wn * 32 + j * 16 + il;
#pragma unroll
                for (int r = 0; r < 8; ++r)
                    acch[(rowb + r) * HD + col] = acc[i][j][r];
            }
        __threadfence();
        __syncthreads();

        // ---- per-column forward substitution (thread d owns column d) ----
        if (tid < HD) {
            const int d = tid;
            float fu[64];
            const unsigned short* vrow = vh + (size_t)I * 64 * HD;
            for (int t = 0; t < 64; ++t) {
                float s = bf2f(vrow[t * HD + d]) - acch[t * HD + d];
                for (int ss = 0; ss < t; ++ss)
                    s -= sS[t * LDSS + ss] * fu[ss];
                fu[t] = s;
            }
            unsigned short* urow = uh + (size_t)I * 64 * HD;
            for (int t = 0; t < 64; ++t)
                urow[t * HD + d] = f2bf(fu[t]);
        }
        __threadfence();
        __syncthreads();
    }
}

// ---------------------------------------------------------------------------
// Stage D: causal flash attention over u, one workgroup per (row-block, head).
// ---------------------------------------------------------------------------
__global__ __launch_bounds__(256) void flash_kernel(
    const unsigned short* __restrict__ qb, const unsigned short* __restrict__ kb,
    const unsigned short* __restrict__ ub, unsigned short* __restrict__ ob)
{
    __shared__ unsigned short sQ[64 * LDQ];
    __shared__ unsigned short sK[64 * LDQ];   // overlaid by f32 S tile
    __shared__ unsigned short sU[128 * LDP];
    __shared__ unsigned short sP[64 * LDP];
    __shared__ float sM[64], sL[64], sC[64];
    float* sS = (float*)sK;

    const int I    = blockIdx.x;   // row block
    const int h    = blockIdx.y;   // head
    const int tid  = threadIdx.x;
    const int lane = tid & 31;
    const int wave = tid >> 5;
    const int wm   = wave >> 2;
    const int wn   = wave & 3;
    const int halfl = lane >> 4, il = lane & 15;

    const unsigned short* qh = qb + (size_t)h * T_DIM * HD;
    const unsigned short* kh = kb + (size_t)h * T_DIM * HD;
    const unsigned short* uh = ub + (size_t)h * T_DIM * HD;

    load_tile64(sQ, qh + (size_t)I * 64 * HD, tid);
    if (tid < 64) { sM[tid] = -3.0e38f; sL[tid] = 0.0f; }

    v8f oacc[2][2];
#pragma unroll
    for (int i = 0; i < 2; ++i)
#pragma unroll
        for (int j = 0; j < 2; ++j) oacc[i][j] = vzero8();
    tile_fence();
    __syncthreads();

    for (int J = 0; J <= I; ++J) {
        load_tile64(sK, kh + (size_t)J * 64 * HD, tid);
        load_u_trans(sU, uh + (size_t)J * 64 * HD, tid);
        tile_fence();
        __syncthreads();

        v8f s0 = vzero8(), s1 = vzero8();
#pragma unroll
        for (int ks = 0; ks < HD; ks += 32) {
            FragBF a0 = load_frag_a(sQ, LDQ, wm * 32,      ks, lane);
            FragBF a1 = load_frag_a(sQ, LDQ, wm * 32 + 16, ks, lane);
            FragBF bk = load_frag_b(sK, LDQ, wn * 16,      ks, lane);
            s0 = wmma_bf16(a0, bk, s0);
            s1 = wmma_bf16(a1, bk, s1);
        }
        __syncthreads();   // everyone done reading sK; overlay with f32 S

        // masked scaled scores into sS
#pragma unroll
        for (int jj = 0; jj < 2; ++jj) {
            v8f& s = jj ? s1 : s0;
            int rowb = wm * 32 + jj * 16 + halfl * 8;
            int col  = wn * 16 + il;
#pragma unroll
            for (int r = 0; r < 8; ++r) {
                int row  = rowb + r;
                int gcol = J * 64 + col;
                int grow = I * 64 + row;
                sS[row * LDSS + col] = (gcol <= grow) ? s[r] * SCALE : -3.0e38f;
            }
        }
        __syncthreads();

        // online softmax per row
        if (tid < 64) {
            const int r = tid;
            float mold = sM[r], mx = mold;
            for (int s = 0; s < 64; ++s) mx = fmaxf(mx, sS[r * LDSS + s]);
            float corr = __expf(mold - mx);
            float sum = 0.0f;
            for (int s = 0; s < 64; ++s) {
                float p = __expf(sS[r * LDSS + s] - mx);
                sum += p;
                sP[r * LDP + s] = f2bf(p);
            }
            sL[r] = sL[r] * corr + sum;
            sM[r] = mx;
            sC[r] = corr;
        }
        __syncthreads();

        // rescale accumulators, then O += P @ U_J
#pragma unroll
        for (int i = 0; i < 2; ++i)
#pragma unroll
            for (int j = 0; j < 2; ++j) {
                int rowb = wm * 32 + i * 16 + halfl * 8;
#pragma unroll
                for (int r = 0; r < 8; ++r) oacc[i][j][r] *= sC[rowb + r];
            }
#pragma unroll
        for (int ks = 0; ks < 64; ks += 32) {
            FragBF p0 = load_frag_a(sP, LDP, wm * 32,      ks, lane);
            FragBF p1 = load_frag_a(sP, LDP, wm * 32 + 16, ks, lane);
            FragBF u0 = load_frag_b(sU, LDP, wn * 32,      ks, lane);
            FragBF u1 = load_frag_b(sU, LDP, wn * 32 + 16, ks, lane);
            oacc[0][0] = wmma_bf16(p0, u0, oacc[0][0]);
            oacc[0][1] = wmma_bf16(p0, u1, oacc[0][1]);
            oacc[1][0] = wmma_bf16(p1, u0, oacc[1][0]);
            oacc[1][1] = wmma_bf16(p1, u1, oacc[1][1]);
        }
        __syncthreads();
    }

    // epilogue: o[t, h*128 + d] = oacc / l, as bf16 for the output GEMM
#pragma unroll
    for (int i = 0; i < 2; ++i)
#pragma unroll
        for (int j = 0; j < 2; ++j) {
            int rowb = wm * 32 + i * 16 + halfl * 8;
            int col  = wn * 32 + j * 16 + il;
#pragma unroll
            for (int r = 0; r < 8; ++r) {
                int row = rowb + r;
                float val = oacc[i][j][r] / sL[row];
                int gt = I * 64 + row;
                ob[(size_t)gt * HID + h * HD + col] = f2bf(val);
            }
        }
}

// ---------------------------------------------------------------------------
// Host-side orchestration
// ---------------------------------------------------------------------------
extern "C" void kernel_launch(void* const* d_in, const int* in_sizes, int n_in,
                              void* d_out, int out_size, void* d_ws, size_t ws_size,
                              hipStream_t stream) {
    (void)in_sizes; (void)n_in; (void)out_size; (void)ws_size;

    const float* x    = (const float*)d_in[0];
    const float* Wq   = (const float*)d_in[1];
    const float* Wk   = (const float*)d_in[2];
    const float* Wv   = (const float*)d_in[3];
    const float* Wb   = (const float*)d_in[4];
    const float* bias = (const float*)d_in[5];
    const float* Wo   = (const float*)d_in[6];
    float* out = (float*)d_out;

    const size_t szHTD = (size_t)NHEAD * T_DIM * HD;   // 4M elements
    unsigned short* qb  = (unsigned short*)d_ws;
    unsigned short* kb  = qb  + szHTD;
    unsigned short* vb  = kb  + szHTD;
    unsigned short* ub  = vb  + szHTD;
    unsigned short* ob  = ub  + szHTD;
    unsigned short* xb  = ob  + szHTD;
    unsigned short* wqb = xb  + szHTD;
    unsigned short* wkb = wqb + szHTD;
    unsigned short* wvb = wkb + szHTD;
    unsigned short* wob = wvb + szHTD;
    float* beta = (float*)(wob + szHTD);
    float* accg = beta + (size_t)NHEAD * T_DIM;

    // stage 0: f32 -> bf16 pre-conversion (x and the four big weights)
    const int cvtBlocks = (T_DIM * HID) / (256 * 4);
    cvt_bf16_kernel<<<cvtBlocks, 256, 0, stream>>>(x,  xb);
    cvt_bf16_kernel<<<cvtBlocks, 256, 0, stream>>>(Wq, wqb);
    cvt_bf16_kernel<<<cvtBlocks, 256, 0, stream>>>(Wk, wkb);
    cvt_bf16_kernel<<<cvtBlocks, 256, 0, stream>>>(Wv, wvb);
    cvt_bf16_kernel<<<cvtBlocks, 256, 0, stream>>>(Wo, wob);

    dim3 gg(T_DIM / 128, HID / 128);

    // stage A: q, k, v projections (bf16 head-major)
    gemm_nt_kernel<1><<<gg, 256, 0, stream>>>(xb, wqb, nullptr, qb, T_DIM, HID, HID);
    gemm_nt_kernel<1><<<gg, 256, 0, stream>>>(xb, wkb, nullptr, kb, T_DIM, HID, HID);
    gemm_nt_kernel<1><<<gg, 256, 0, stream>>>(xb, wvb, nullptr, vb, T_DIM, HID, HID);

    // stage B: beta
    beta_kernel<<<(T_DIM * NHEAD) / 256, 256, 0, stream>>>(x, Wb, bias, beta);

    // stage C: delta-rule solve (one WG per head, sequential over row blocks)
    delta_solve_kernel<<<NHEAD, 256, 0, stream>>>(qb, kb, vb, beta, ub, accg);

    // stage D: causal flash attention over u
    flash_kernel<<<dim3(T_DIM / 64, NHEAD), 256, 0, stream>>>(qb, kb, ub, ob);

    // stage E: output projection -> f32
    gemm_nt_kernel<0><<<gg, 256, 0, stream>>>(ob, wob, out, nullptr, T_DIM, HID, HID);
}
